// ConditionalDetrDecoderCrossAttention_10453950398827
// MI455X (gfx1250) — matrix operand order repaired and checked by reading, not compiled, not run on hardware
//
#include <hip/hip_runtime.h>
#include <hip/hip_bf16.h>
#include <cstddef>
#include <cstdint>

// ------------------------------------------------------------------
// ConditionalDETR decoder cross-attention, MI455X (gfx1250, wave32).
// Memory-bound (attn_weights output alone = 629 MB fp32); compute via
// v_wmma_f32_16x16x32_bf16 (f32 accumulate). V tiles are staged into
// LDS with async global->LDS DMA overlapped with softmax normalize.
// ------------------------------------------------------------------

#define B_   32
#define Q_   300
#define QP_  304      // Q padded to 16
#define K_   2048
#define HID_ 256
#define NH_  8
#define HD_  32       // per-head dim (concat head dim = 64)

typedef __attribute__((ext_vector_type(16))) __bf16 v16bf;
typedef __attribute__((ext_vector_type(8)))  float  v8f;

struct __align__(16) U4 { unsigned int x, y, z, w; };

union BF16x16 {
    v16bf          v;
    unsigned short u[16];
    unsigned int   d[8];
    U4             q[2];
};

__device__ __forceinline__ unsigned short f2bf(float f) {
    unsigned int x = __float_as_uint(f);
    x += 0x7FFFu + ((x >> 16) & 1u);          // round-to-nearest-even
    return (unsigned short)(x >> 16);
}

// Packed f32x2 -> bf16x2 (one VALU op on gfx1250 when available).
__device__ __forceinline__ unsigned int pk_bf16(float lo, float hi) {
#if __has_builtin(__builtin_amdgcn_cvt_pk_bf16_f32)
    auto r = __builtin_amdgcn_cvt_pk_bf16_f32(lo, hi);
    unsigned int u;
    __builtin_memcpy(&u, &r, 4);
    return u;
#else
    return (unsigned int)f2bf(lo) | ((unsigned int)f2bf(hi) << 16);
#endif
}

__device__ __forceinline__ v8f wmma_bf16(const BF16x16& a, const BF16x16& b, v8f c) {
    return __builtin_amdgcn_wmma_f32_16x16x32_bf16(false, a.v, false, b.v,
                                                   (short)0, c, false, false);
}

// ---- async global -> LDS (16B per lane), ASYNCcnt-tracked --------
#if __has_builtin(__builtin_amdgcn_global_load_async_to_lds_b128)
#define HAVE_ASYNC_LDS 1
#endif

typedef int v4i_vs __attribute__((vector_size(16)));

__device__ __forceinline__ void async_copy16(void* g, void* l) {
#ifdef HAVE_ASYNC_LDS
    __builtin_amdgcn_global_load_async_to_lds_b128(
        (__attribute__((address_space(1))) v4i_vs*)g,
        (__attribute__((address_space(3))) v4i_vs*)l, 0, 0);
#else
    *(U4*)l = *(const U4*)g;                 // synchronous fallback
#endif
}
__device__ __forceinline__ void wait_async() {
#ifdef HAVE_ASYNC_LDS
#if __has_builtin(__builtin_amdgcn_s_wait_asynccnt)
    __builtin_amdgcn_s_wait_asynccnt(0);
#else
    asm volatile("s_wait_asynccnt 0x0" ::: "memory");
#endif
#endif
}

// A fragment (16x32 bf16) from an fp32 row, converting on the fly.
__device__ __forceinline__ void loadA_f32(BF16x16& a, const float* row, int kk, int kb) {
#pragma unroll
    for (int j = 0; j < 4; ++j) {
        a.d[j]     = pk_bf16(row[kk + kb + 2 * j],      row[kk + kb + 2 * j + 1]);
        a.d[4 + j] = pk_bf16(row[kk + kb + 16 + 2 * j], row[kk + kb + 16 + 2 * j + 1]);
    }
}
__device__ __forceinline__ void zeroA(BF16x16& a) {
#pragma unroll
    for (int j = 0; j < 8; ++j) a.d[j] = 0;
}
// A fragment from a bf16 row (two contiguous 16B loads).
__device__ __forceinline__ void loadA_bf16(BF16x16& a, const unsigned short* p) {
    a.q[0] = *(const U4*)p;          // K = kb + 0..7
    a.q[1] = *(const U4*)(p + 16);   // K = kb + 16..23
}
// B fragment: lane holds column N = lane%16, K = 8h+0..7 and 16+8h+0..7 (strided by ld).
__device__ __forceinline__ void loadB_bf16(BF16x16& b, const unsigned short* p, int ld) {
    const int l = threadIdx.x & 31, n = l & 15, hk = (l >> 4) * 8;
#pragma unroll
    for (int j = 0; j < 8; ++j) {
        b.u[j]     = p[(size_t)(hk + j) * ld + n];
        b.u[8 + j] = p[(size_t)(16 + hk + j) * ld + n];
    }
}

// ------------------------------------------------------------------
// Kernel 0: transpose weights to bf16 [in][out]; fold biases.
// wT order: 0 qc, 1 qp, 2 qs, 3 kc, 4 kp, 5 v, 6 o
// bias order: [qb=qc+qp, qsb, kb=kc+kp, kpb, vb, ob] * 256
// ------------------------------------------------------------------
__global__ void prep_kernel(const float* qc_w, const float* qp_w, const float* qs_w,
                            const float* kc_w, const float* kp_w, const float* v_w,
                            const float* o_w,
                            const float* qc_b, const float* qp_b, const float* qs_b,
                            const float* kc_b, const float* kp_b, const float* v_b,
                            const float* o_b,
                            unsigned short* wT, float* bias) {
    int idx = blockIdx.x * 256 + threadIdx.x;
    const int NW = 7 * 65536;
    if (idx < NW) {
        int w = idx >> 16, rem = idx & 65535;
        int o = rem >> 8, i = rem & 255;
        const float* src;
        switch (w) {
            case 0: src = qc_w; break;  case 1: src = qp_w; break;
            case 2: src = qs_w; break;  case 3: src = kc_w; break;
            case 4: src = kp_w; break;  case 5: src = v_w;  break;
            default: src = o_w;
        }
        wT[(size_t)w * 65536 + (size_t)i * 256 + o] = f2bf(src[(size_t)o * 256 + i]);
    } else if (idx < NW + 1536) {
        int r = idx - NW, which = r >> 8, j = r & 255;
        float v;
        switch (which) {
            case 0: v = qc_b[j] + qp_b[j]; break;
            case 1: v = qs_b[j];           break;
            case 2: v = kc_b[j] + kp_b[j]; break;
            case 3: v = kp_b[j];           break;
            case 4: v = v_b[j];            break;
            default: v = o_b[j];
        }
        bias[r] = v;
    }
}

// ------------------------------------------------------------------
// Kernel 1: q_full bf16 [b,h,QP_,64]. Virtual GEMM N=512:
//   n<256 : hs@qcT + qpe@qpT + qb   -> cols 0..31 of head n/32
//   n>=256: qse@qsT + qsb           -> cols 32..63 of head (n-256)/32
// ------------------------------------------------------------------
__global__ __launch_bounds__(32) void qproj_kernel(const float* hs, const float* qpe,
                                                   const float* qse,
                                                   const unsigned short* wT,
                                                   const float* bias,
                                                   unsigned short* qf) {
    const int lane = threadIdx.x;
    const int b = blockIdx.z, mt = blockIdx.y, n0 = blockIdx.x * 16;
    const int arow = lane & 15, hi = lane >> 4, kb = hi * 8;
    const int q0 = mt * 16, q = q0 + arow;
    const bool qv = (q < Q_);

    const float* r0;
    const float* r1 = nullptr;
    const unsigned short* w0;
    const unsigned short* w1 = nullptr;
    int ncol;
    if (n0 < 256) {
        r0 = hs  + ((size_t)b * Q_ + q) * HID_;
        r1 = qpe + ((size_t)b * Q_ + q) * HID_;
        w0 = wT + 0 * 65536; w1 = wT + 1 * 65536; ncol = n0;
    } else {
        r0 = qse + ((size_t)b * Q_ + q) * HID_;
        w0 = wT + 2 * 65536; ncol = n0 - 256;
    }

    v8f acc = {};
    for (int kk = 0; kk < HID_; kk += 32) {
        BF16x16 a, bb;
        if (qv) loadA_f32(a, r0, kk, kb); else zeroA(a);
        loadB_bf16(bb, w0 + (size_t)kk * 256 + ncol, 256);
        acc = wmma_bf16(a, bb, acc);
        if (w1) {
            BF16x16 a2, bb2;
            if (qv) loadA_f32(a2, r1, kk, kb); else zeroA(a2);
            loadB_bf16(bb2, w1 + (size_t)kk * 256 + ncol, 256);
            acc = wmma_bf16(a2, bb2, acc);
        }
    }

    const int n = arow;
    const float bia = bias[n0 + n];
    int nn = (n0 < 256) ? (n0 + n) : (n0 - 256 + n);
    int h  = nn >> 5;
    int j  = (n0 < 256) ? (nn & 31) : (32 + (nn & 31));
    unsigned short* dst = qf + (((size_t)b * NH_ + h) * QP_) * 64;
#pragma unroll
    for (int r = 0; r < 8; ++r) {
        int M = r + hi * 8;
        dst[(size_t)(q0 + M) * 64 + j] = f2bf(acc[r] + bia);
    }
}

// ------------------------------------------------------------------
// Kernel 2: k_full transposed bf16 [b,h,64,K_] and v bf16 [b,h,K_,32].
// Virtual GEMM N=768:
//   n<256  : ehs@kcT + epe@kpT + kb -> kfT cols 0..31
//   n<512  : epe@kpT + kpb          -> kfT cols 32..63
//   else   : ehs@vT  + vb           -> vws
// ------------------------------------------------------------------
__global__ __launch_bounds__(32) void kproj_kernel(const float* ehs, const float* epe,
                                                   const unsigned short* wT,
                                                   const float* bias,
                                                   unsigned short* kfT,
                                                   unsigned short* vws) {
    const int lane = threadIdx.x;
    const int b = blockIdx.z, mt = blockIdx.y, n0 = blockIdx.x * 16;
    const int arow = lane & 15, hi = lane >> 4, kb = hi * 8;
    const int k0m = mt * 16, krow = k0m + arow;

    const float* r0;
    const float* r1 = nullptr;
    const unsigned short* w0;
    const unsigned short* w1 = nullptr;
    int ncol, sec;
    const size_t rowoff = ((size_t)b * K_ + krow) * HID_;
    if (n0 < 256)       { sec = 0; r0 = ehs + rowoff; r1 = epe + rowoff;
                          w0 = wT + 3 * 65536; w1 = wT + 4 * 65536; ncol = n0; }
    else if (n0 < 512)  { sec = 1; r0 = epe + rowoff; w0 = wT + 4 * 65536; ncol = n0 - 256; }
    else                { sec = 2; r0 = ehs + rowoff; w0 = wT + 5 * 65536; ncol = n0 - 512; }

    v8f acc = {};
    for (int kk = 0; kk < HID_; kk += 32) {
        BF16x16 a, bb;
        loadA_f32(a, r0, kk, kb);
        loadB_bf16(bb, w0 + (size_t)kk * 256 + ncol, 256);
        acc = wmma_bf16(a, bb, acc);
        if (w1) {
            BF16x16 a2, bb2;
            loadA_f32(a2, r1, kk, kb);
            loadB_bf16(bb2, w1 + (size_t)kk * 256 + ncol, 256);
            acc = wmma_bf16(a2, bb2, acc);
        }
    }

    const int n = arow;
    const float bia = bias[512 + n0 + n];
    if (sec < 2) {
        int nn = (sec == 0) ? (n0 + n) : (n0 - 256 + n);
        int h  = nn >> 5;
        int j  = (sec == 0) ? (nn & 31) : (32 + (nn & 31));
        unsigned short* dst = kfT + (((size_t)b * NH_ + h) * 64 + j) * K_;
#pragma unroll
        for (int r = 0; r < 8; ++r)
            dst[k0m + r + hi * 8] = f2bf(acc[r] + bia);
    } else {
        int nn = n0 - 512 + n;
        int h = nn >> 5, j = nn & 31;
        unsigned short* dst = vws + ((size_t)b * NH_ + h) * (size_t)K_ * 32;
#pragma unroll
        for (int r = 0; r < 8; ++r)
            dst[(size_t)(k0m + r + hi * 8) * 32 + j] = f2bf(acc[r] + bia);
    }
}

// ------------------------------------------------------------------
// Kernel 3: attention. One 4-wave workgroup per (qtile, h, b).
// Phase 1: scores in 256-col LDS chunks -> raw scaled scores to d_out,
//          running row (max, sum) with rescaling.
// Phase 2: async-stage V chunk into LDS (overlapped with in-place
//          normalize of d_out), then WMMA P@V from LDS; cross-wave
//          reduce -> bf16 attn_out scratch.
// ------------------------------------------------------------------
#define CHUNK 256

__global__ __launch_bounds__(128) void attn_kernel(const unsigned short* qf,
                                                   const unsigned short* kfT,
                                                   const unsigned short* vws,
                                                   unsigned short* ao,
                                                   float* out) {
    __shared__ float          sS[16 * CHUNK];          // 16 KB raw scores
    __shared__ unsigned short sP[16 * CHUNK];          //  8 KB bf16 probs
    __shared__ unsigned short sV[CHUNK * 32];          // 16 KB staged V chunk
    __shared__ float          sMax[16], sSum[16], sRed[128];
    __shared__ float          sAcc[4 * 16 * 32];       //  8 KB cross-wave PV

    const int tid = threadIdx.x, lane = tid & 31, wv = tid >> 5;
    const int qt = blockIdx.x, h = blockIdx.y, b = blockIdx.z;
    const int bh = b * NH_ + h;
    const int q0 = qt * 16;
    const int arow = lane & 15, hi = lane >> 4, kb = hi * 8;

    const size_t AW_OFF = (size_t)B_ * Q_ * HID_;
    float* aw = out + AW_OFF + (size_t)bh * Q_ * K_;
    const unsigned short* qbase = qf  + ((size_t)bh * QP_ + q0) * 64;
    const unsigned short* kbase = kfT + ((size_t)bh * 64) * K_;
    const unsigned short* vbase = vws + ((size_t)bh * K_) * 32;

    if (tid < 16) { sMax[tid] = -3.4e38f; sSum[tid] = 0.f; }
    __syncthreads();

    // ---------------- Phase 1: scores + running softmax stats ----------------
    for (int c = 0; c < K_ / CHUNK; ++c) {
        const int kc0 = c * CHUNK;
        for (int i = 0; i < (CHUNK / 16) / 4; ++i) {       // tiles per wave
            const int t = wv + 4 * i;
            const int ncol = kc0 + t * 16;
            v8f acc = {};
            for (int kk = 0; kk < 64; kk += 32) {
                BF16x16 a, bb;
                loadA_bf16(a, qbase + (size_t)arow * 64 + kk + kb);
                loadB_bf16(bb, kbase + (size_t)kk * K_ + ncol, K_);
                acc = wmma_bf16(a, bb, acc);
            }
#pragma unroll
            for (int r = 0; r < 8; ++r) {
                int M = r + hi * 8;
                float v = acc[r] * 0.125f;                 // (2*HD)^-0.5
                sS[M * CHUNK + t * 16 + arow] = v;
                int qq = q0 + M;
                if (qq < Q_) aw[(size_t)qq * K_ + ncol + arow] = v;
            }
        }
        __threadfence_block();
        __syncthreads();

        {   // row stats: 8 threads per row
            int row = tid >> 3, seg = tid & 7;
            const float* sr = sS + row * CHUNK + seg * (CHUNK / 8);
            float lm = -3.4e38f;
            for (int j = 0; j < CHUNK / 8; ++j) lm = fmaxf(lm, sr[j]);
            sRed[tid] = lm;
            __syncthreads();
            if (tid < 16) {
                float cm = sRed[tid * 8];
                for (int j = 1; j < 8; ++j) cm = fmaxf(cm, sRed[tid * 8 + j]);
                float om = sMax[tid];
                float nm = fmaxf(om, cm);
                sSum[tid] *= __expf(om - nm);
                sMax[tid]  = nm;
            }
            __syncthreads();
            float nm = sMax[row];
            float ls = 0.f;
            for (int j = 0; j < CHUNK / 8; ++j) ls += __expf(sr[j] - nm);
            sRed[tid] = ls;
            __syncthreads();
            if (tid < 16) {
                float s = 0.f;
                for (int j = 0; j < 8; ++j) s += sRed[tid * 8 + j];
                sSum[tid] += s;
            }
            __syncthreads();
        }
    }

    // ---------------- Phase 2: normalize + P@V ----------------
    v8f acc0 = {}, acc1 = {};
    for (int c = 0; c < K_ / CHUNK; ++c) {
        const int kc0 = c * CHUNK;

        // kick off async DMA of this V chunk (CHUNK x 32 bf16) into LDS;
        // overlaps with the exp/normalize global traffic below.
        for (int i = tid; i < CHUNK * 4; i += 128)          // 16B units
            async_copy16((void*)(vbase + (size_t)kc0 * 32 + (size_t)i * 8),
                         (void*)(sV + (size_t)i * 8));

        for (int i = tid; i < 16 * CHUNK; i += 128) {
            int row = i / CHUNK, col = i % CHUNK;
            int qq = q0 + row;
            float p = 0.f;
            if (qq < Q_) {
                size_t gi = (size_t)qq * K_ + kc0 + col;
                float s = aw[gi];
                p = __expf(s - sMax[row]) / sSum[row];
                aw[gi] = p;                                 // normalized attn_weights
            }
            sP[row * CHUNK + col] = f2bf(p);
        }
        wait_async();
        __syncthreads();

        for (int ks = 0; ks < CHUNK / 128; ++ks) {          // 64 K-cols per wave
            const int kc = wv * (CHUNK / 4) + ks * 32;
            BF16x16 a, b0, b1;
            const unsigned short* ap = sP + arow * CHUNK + kc + kb;
#pragma unroll
            for (int j = 0; j < 8; ++j) { a.u[j] = ap[j]; a.u[8 + j] = ap[16 + j]; }
            const unsigned short* vp = sV + (size_t)kc * 32;
            loadB_bf16(b0, vp, 32);
            loadB_bf16(b1, vp + 16, 32);
            acc0 = wmma_bf16(a, b0, acc0);
            acc1 = wmma_bf16(a, b1, acc1);
        }
        __syncthreads();
    }

    // cross-wave reduction of the 16x32 partials
#pragma unroll
    for (int r = 0; r < 8; ++r) {
        int M = r + hi * 8;
        sAcc[(wv * 16 + M) * 32 + arow]      = acc0[r];
        sAcc[(wv * 16 + M) * 32 + 16 + arow] = acc1[r];
    }
    __syncthreads();
    for (int i = tid; i < 16 * 32; i += 128) {
        int M = i >> 5, d = i & 31;
        float v = sAcc[M * 32 + d] + sAcc[(16 + M) * 32 + d] +
                  sAcc[(32 + M) * 32 + d] + sAcc[(48 + M) * 32 + d];
        ao[((size_t)b * QP_ + q0 + M) * HID_ + h * HD_ + d] = f2bf(v);
    }
}

// ------------------------------------------------------------------
// Kernel 4: output projection attn_out @ o_w^T + o_b -> d_out (fp32).
// ------------------------------------------------------------------
__global__ __launch_bounds__(32) void oproj_kernel(const unsigned short* ao,
                                                   const unsigned short* wT,
                                                   const float* bias, float* out) {
    const int lane = threadIdx.x;
    const int b = blockIdx.z, mt = blockIdx.y, n0 = blockIdx.x * 16;
    const int arow = lane & 15, hi = lane >> 4, kb = hi * 8;
    const int q0 = mt * 16, q = q0 + arow;
    const unsigned short* oT = wT + 6 * 65536;
    const unsigned short* r0 = ao + ((size_t)b * QP_ + q) * HID_;

    v8f acc = {};
    for (int kk = 0; kk < HID_; kk += 32) {
        BF16x16 a, bb;
        loadA_bf16(a, r0 + kk + kb);
        loadB_bf16(bb, oT + (size_t)kk * 256 + n0, 256);
        acc = wmma_bf16(a, bb, acc);
    }
    const int n = arow;
    const float bia = bias[1280 + n0 + n];
#pragma unroll
    for (int r = 0; r < 8; ++r) {
        int M = r + hi * 8, qq = q0 + M;
        if (qq < Q_)
            out[((size_t)b * Q_ + qq) * HID_ + n0 + n] = acc[r] + bia;
    }
}

// ------------------------------------------------------------------
// Launch
// ------------------------------------------------------------------
extern "C" void kernel_launch(void* const* d_in, const int* in_sizes, int n_in,
                              void* d_out, int out_size, void* d_ws, size_t ws_size,
                              hipStream_t stream) {
    (void)in_sizes; (void)n_in; (void)out_size; (void)ws_size;
    const float* hs   = (const float*)d_in[0];
    const float* ehs  = (const float*)d_in[1];
    const float* qse  = (const float*)d_in[2];
    const float* epe  = (const float*)d_in[3];
    const float* qpe  = (const float*)d_in[4];
    const float* qc_w = (const float*)d_in[5];  const float* qc_b = (const float*)d_in[6];
    const float* qp_w = (const float*)d_in[7];  const float* qp_b = (const float*)d_in[8];
    const float* kc_w = (const float*)d_in[9];  const float* kc_b = (const float*)d_in[10];
    const float* kp_w = (const float*)d_in[11]; const float* kp_b = (const float*)d_in[12];
    const float* v_w  = (const float*)d_in[13]; const float* v_b  = (const float*)d_in[14];
    const float* qs_w = (const float*)d_in[15]; const float* qs_b = (const float*)d_in[16];
    const float* o_w  = (const float*)d_in[17]; const float* o_b  = (const float*)d_in[18];

    // workspace layout (bytes)
    constexpr size_t OFF_BIAS = (size_t)7 * 65536 * 2;                     // 917504
    constexpr size_t OFF_QF   = OFF_BIAS + (size_t)1536 * 4;               // 923648
    constexpr size_t OFF_KFT  = OFF_QF  + (size_t)B_ * NH_ * QP_ * 64 * 2; // +9.96 MB
    constexpr size_t OFF_VWS  = OFF_KFT + (size_t)B_ * NH_ * 64 * K_ * 2;  // +67 MB
    constexpr size_t OFF_AO   = OFF_VWS + (size_t)B_ * NH_ * K_ * 32 * 2;  // +33.5 MB

    char* ws = (char*)d_ws;
    unsigned short* wT   = (unsigned short*)(ws);
    float*          bias = (float*)(ws + OFF_BIAS);
    unsigned short* qf   = (unsigned short*)(ws + OFF_QF);
    unsigned short* kfT  = (unsigned short*)(ws + OFF_KFT);
    unsigned short* vws  = (unsigned short*)(ws + OFF_VWS);
    unsigned short* ao   = (unsigned short*)(ws + OFF_AO);
    float* out = (float*)d_out;

    prep_kernel<<<dim3(1798), dim3(256), 0, stream>>>(
        qc_w, qp_w, qs_w, kc_w, kp_w, v_w, o_w,
        qc_b, qp_b, qs_b, kc_b, kp_b, v_b, o_b, wT, bias);

    qproj_kernel<<<dim3(32, QP_ / 16, B_), dim3(32), 0, stream>>>(hs, qpe, qse, wT, bias, qf);
    kproj_kernel<<<dim3(48, K_ / 16, B_), dim3(32), 0, stream>>>(ehs, epe, wT, bias, kfT, vws);
    attn_kernel<<<dim3(QP_ / 16, NH_, B_), dim3(128), 0, stream>>>(qf, kfT, vws, ao, out);
    oproj_kernel<<<dim3(HID_ / 16, QP_ / 16, B_), dim3(32), 0, stream>>>(ao, wT, bias, out);
}
